// GatNet_42322607735199
// MI455X (gfx1250) — compile-verified, matrix operand order checked by reading, not run on hardware
//
#include <hip/hip_runtime.h>
#include <hip/hip_bf16.h>

typedef __attribute__((ext_vector_type(16))) __bf16 v16bf;
typedef __attribute__((ext_vector_type(8)))  __bf16 v8bf;
typedef __attribute__((ext_vector_type(8)))  float  v8f;

#define F_IN   256
#define HEADS  8
#define C1     8
#define HC     64      // HEADS*C1
#define NCLS   16
#define NEG_SLOPE 0.2f

// ---- order-preserving float<->uint encoding for atomicMax-based segment_max ----
__device__ __forceinline__ unsigned enc_f32(float f) {
  unsigned u = __float_as_uint(f);
  return (u & 0x80000000u) ? ~u : (u | 0x80000000u);
}
__device__ __forceinline__ float dec_f32(unsigned u) {
  return (u & 0x80000000u) ? __uint_as_float(u ^ 0x80000000u) : __uint_as_float(~u);
}
__device__ __forceinline__ float leaky(float x) { return x > 0.f ? x : NEG_SLOPE * x; }

// async global->LDS copy of one 16-byte chunk (CDNA5 GLOBAL_LOAD_ASYNC_TO_LDS_B128)
__device__ __forceinline__ void async_copy_b128(const void* gptr, void* lptr) {
  unsigned lds_addr  = (unsigned)(unsigned long long)lptr;   // low 32 bits = LDS offset
  unsigned long long gaddr = (unsigned long long)gptr;
  asm volatile("global_load_async_to_lds_b128 %0, %1, off"
               :: "v"(lds_addr), "v"(gaddr) : "memory");
}
__device__ __forceinline__ void async_wait() {
  asm volatile("s_wait_asynccnt 0x0" ::: "memory");
}

// ---------------- zero-init for atomic accumulators ----------------
__global__ void zero_kernel(float* __restrict__ p, size_t n) {
  size_t i = (size_t)blockIdx.x * blockDim.x + threadIdx.x;
  if (i < n) p[i] = 0.f;
}

// ---------------- pre-swizzle W1/W2 into per-lane-contiguous bf16 fragments ----
// W1sw fragment layout: idx = (((s*4+t)*32)+lane)*16 + i
//   k = s*32 + (lane>>4)*8 + (i<8 ? i : i+8),  n = t*16 + (lane&15)
// W2sw fragment layout: idx = ((s*32)+lane)*16 + i,  n = lane&15
__global__ void prep_w_kernel(const float* __restrict__ W1, const float* __restrict__ W2,
                              __bf16* __restrict__ W1sw, __bf16* __restrict__ W2sw) {
  int tid = blockIdx.x * blockDim.x + threadIdx.x;
  int stride = gridDim.x * blockDim.x;
  for (int idx = tid; idx < 8 * 4 * 32 * 16; idx += stride) {
    int i = idx & 15, lane = (idx >> 4) & 31, t = (idx >> 9) & 3, s = idx >> 11;
    int k = s * 32 + (lane >> 4) * 8 + (i < 8 ? i : i + 8);
    int n = t * 16 + (lane & 15);
    W1sw[idx] = (__bf16)W1[k * HC + n];
  }
  for (int idx = tid; idx < 2 * 32 * 16; idx += stride) {
    int i = idx & 15, lane = (idx >> 4) & 31, s = idx >> 9;
    int k = s * 32 + (lane >> 4) * 8 + (i < 8 ? i : i + 8);
    int n = lane & 15;
    W2sw[idx] = (__bf16)W2[k * NCLS + n];
  }
}

// ---------------- Layer 1 GEMM: h1 = x @ W1 (bf16 WMMA, f32 acc) + attn logits ----
__global__ __launch_bounds__(128)
void gemm1_kernel(const float* __restrict__ x, const __bf16* __restrict__ W1sw,
                  const float* __restrict__ a1s, const float* __restrict__ a1d,
                  float* __restrict__ h1, float* __restrict__ al1s,
                  float* __restrict__ al1d, int N)
{
  __shared__ __bf16 Wsw[8 * 4 * 32 * 16];   // 32 KB swizzled weight fragments
  __shared__ float  Hs[64][HC + 1];         // result staging for logits

  const int tid = threadIdx.x;
  const int nb  = blockIdx.x * 64;

  // bulk async copy of the swizzled weight tile into LDS (16 B chunks)
  for (int c = tid; c < (8 * 4 * 32 * 16) / 8; c += 128)
    async_copy_b128(&W1sw[c * 8], &Wsw[c * 8]);
  async_wait();
  __syncthreads();

  const int lane  = tid & 31;
  const int wave  = tid >> 5;
  const int mrow  = lane & 15;
  const int khalf = lane >> 4;       // 0: K 0-7/16-23 ; 1: K 8-15/24-31
  const int k0    = khalf * 8;
  const int rbase = wave * 16;

  int node = nb + rbase + mrow;
  if (node >= N) node = N - 1;       // clamp; padded rows masked at store
  const float* __restrict__ xrow = x + (size_t)node * F_IN;

  v8f acc[4] = {};
  #pragma unroll
  for (int s = 0; s < 8; ++s) {
    const int kb = s * 32;
    v16bf a;
    #pragma unroll
    for (int i = 0; i < 16; ++i) {
      int k = kb + k0 + (i < 8 ? i : i + 8);
      a[i] = (__bf16)xrow[k];
    }
    #pragma unroll
    for (int t = 0; t < 4; ++t) {
      const v16bf b = *(const v16bf*)&Wsw[(((s << 2) | t) * 32 + lane) * 16];
      acc[t] = __builtin_amdgcn_wmma_f32_16x16x32_bf16(
          false, a, false, b, (short)0, acc[t], false, false);
    }
  }

  // D layout: VGPR r -> M = r + khalf*8, N = mrow (within 16-wide tile t)
  #pragma unroll
  for (int t = 0; t < 4; ++t)
    #pragma unroll
    for (int r = 0; r < 8; ++r)
      Hs[rbase + r + khalf * 8][t * 16 + mrow] = acc[t][r];
  __syncthreads();

  for (int idx = tid; idx < 64 * HC; idx += 128) {
    int r = idx >> 6, c = idx & 63;
    int nd = nb + r;
    if (nd < N) h1[(size_t)nd * HC + c] = Hs[r][c];
  }
  for (int p = tid; p < 64 * HEADS; p += 128) {
    int r = p >> 3, hd = p & 7;
    int nd = nb + r;
    if (nd < N) {
      float ss = 0.f, sd = 0.f;
      #pragma unroll
      for (int c = 0; c < C1; ++c) {
        float v = Hs[r][hd * C1 + c];
        ss += v * a1s[hd * C1 + c];
        sd += v * a1d[hd * C1 + c];
      }
      al1s[(size_t)nd * HEADS + hd] = ss;
      al1d[(size_t)nd * HEADS + hd] = sd;
    }
  }
}

// ---------------- Layer 1 edge passes ----------------
__global__ void attmax1_kernel(const int* __restrict__ ei, const float* __restrict__ als,
                               const float* __restrict__ ald, unsigned* __restrict__ m1,
                               int E, int Etot) {
  int e = blockIdx.x * blockDim.x + threadIdx.x;
  if (e >= Etot) return;
  int src = (e < E) ? ei[e]     : (e - E);
  int dst = (e < E) ? ei[E + e] : (e - E);
  #pragma unroll
  for (int h = 0; h < HEADS; ++h) {
    float v = leaky(als[(size_t)src * HEADS + h] + ald[(size_t)dst * HEADS + h]);
    atomicMax(&m1[(size_t)dst * HEADS + h], enc_f32(v));
  }
}

__global__ void denom1_kernel(const int* __restrict__ ei, const float* __restrict__ als,
                              const float* __restrict__ ald, const unsigned* __restrict__ m1,
                              float* __restrict__ den1, int E, int Etot) {
  int e = blockIdx.x * blockDim.x + threadIdx.x;
  if (e >= Etot) return;
  int src = (e < E) ? ei[e]     : (e - E);
  int dst = (e < E) ? ei[E + e] : (e - E);
  #pragma unroll
  for (int h = 0; h < HEADS; ++h) {
    float v = leaky(als[(size_t)src * HEADS + h] + ald[(size_t)dst * HEADS + h]);
    float a = __expf(v - dec_f32(m1[(size_t)dst * HEADS + h]));
    atomicAdd(&den1[(size_t)dst * HEADS + h], a);
  }
}

__global__ void agg1_kernel(const int* __restrict__ ei, const float* __restrict__ als,
                            const float* __restrict__ ald, const unsigned* __restrict__ m1,
                            const float* __restrict__ den1, const float* __restrict__ h1,
                            float* __restrict__ agg1, int E, int Etot) {
  long long t = (long long)blockIdx.x * blockDim.x + threadIdx.x;
  if (t >= (long long)Etot * HC) return;
  int e   = (int)(t >> 6);
  int col = (int)(t & 63);
  int hd  = col >> 3;
  int src = (e < E) ? ei[e]     : (e - E);
  int dst = (e < E) ? ei[E + e] : (e - E);
  __builtin_prefetch(&h1[(size_t)src * HC + col], 0, 0);
  float v = leaky(als[(size_t)src * HEADS + hd] + ald[(size_t)dst * HEADS + hd]);
  float alpha = __expf(v - dec_f32(m1[(size_t)dst * HEADS + hd]))
                / den1[(size_t)dst * HEADS + hd];
  atomicAdd(&agg1[(size_t)dst * HC + col], h1[(size_t)src * HC + col] * alpha);
}

// ---------------- elu + Layer 2 GEMM (WMMA) + attn logits ----------------
__global__ __launch_bounds__(128)
void gemm2_kernel(const float* __restrict__ agg1, const float* __restrict__ b1,
                  const __bf16* __restrict__ W2sw, const float* __restrict__ a2s,
                  const float* __restrict__ a2d, float* __restrict__ h2,
                  float* __restrict__ al2s, float* __restrict__ al2d, int N)
{
  __shared__ __bf16 Hs[64][HC];        // elu(h1+b1) in bf16, row-major
  __shared__ __bf16 Wsw[2 * 32 * 16];  // 2 KB swizzled W2 fragments
  __shared__ float  Os[64][NCLS + 1];

  const int tid = threadIdx.x;
  const int nb  = blockIdx.x * 64;

  // async copy of swizzled W2 fragments (16 B per thread covers 2 KB)
  for (int c = tid; c < (2 * 32 * 16) / 8; c += 128)
    async_copy_b128(&W2sw[c * 8], &Wsw[c * 8]);

  for (int idx = tid; idx < 64 * HC; idx += 128) {
    int r = idx >> 6, c = idx & 63;
    int nd = nb + r;
    float v = (nd < N) ? (agg1[(size_t)nd * HC + c] + b1[c]) : 0.f;
    v = (v > 0.f) ? v : expm1f(v);   // elu
    Hs[r][c] = (__bf16)v;
  }
  async_wait();
  __syncthreads();

  const int lane  = tid & 31;
  const int wave  = tid >> 5;
  const int mrow  = lane & 15;
  const int khalf = lane >> 4;
  const int k0    = khalf * 8;
  const int rbase = wave * 16;

  v8f acc = {};
  #pragma unroll
  for (int s = 0; s < 2; ++s) {
    const int kb = s * 32;
    // A fragment: two contiguous 16-byte runs per lane
    const v8bf alo = *(const v8bf*)&Hs[rbase + mrow][kb + k0];
    const v8bf ahi = *(const v8bf*)&Hs[rbase + mrow][kb + k0 + 16];
    v16bf a;
    #pragma unroll
    for (int i = 0; i < 8; ++i) { a[i] = alo[i]; a[i + 8] = ahi[i]; }
    const v16bf b = *(const v16bf*)&Wsw[(s * 32 + lane) * 16];
    acc = __builtin_amdgcn_wmma_f32_16x16x32_bf16(
        false, a, false, b, (short)0, acc, false, false);
  }
  #pragma unroll
  for (int r = 0; r < 8; ++r)
    Os[rbase + r + khalf * 8][mrow] = acc[r];
  __syncthreads();

  for (int r = tid; r < 64; r += 128) {
    int nd = nb + r;
    if (nd < N) {
      float ss = 0.f, sd = 0.f;
      #pragma unroll
      for (int c = 0; c < NCLS; ++c) {
        float v = Os[r][c];
        h2[(size_t)nd * NCLS + c] = v;
        ss += v * a2s[c];
        sd += v * a2d[c];
      }
      al2s[nd] = ss;
      al2d[nd] = sd;
    }
  }
}

// ---------------- Layer 2 edge passes (1 head) ----------------
__global__ void attmax2_kernel(const int* __restrict__ ei, const float* __restrict__ als,
                               const float* __restrict__ ald, unsigned* __restrict__ m2,
                               int E, int Etot) {
  int e = blockIdx.x * blockDim.x + threadIdx.x;
  if (e >= Etot) return;
  int src = (e < E) ? ei[e]     : (e - E);
  int dst = (e < E) ? ei[E + e] : (e - E);
  atomicMax(&m2[dst], enc_f32(leaky(als[src] + ald[dst])));
}

__global__ void denom2_kernel(const int* __restrict__ ei, const float* __restrict__ als,
                              const float* __restrict__ ald, const unsigned* __restrict__ m2,
                              float* __restrict__ den2, int E, int Etot) {
  int e = blockIdx.x * blockDim.x + threadIdx.x;
  if (e >= Etot) return;
  int src = (e < E) ? ei[e]     : (e - E);
  int dst = (e < E) ? ei[E + e] : (e - E);
  atomicAdd(&den2[dst], __expf(leaky(als[src] + ald[dst]) - dec_f32(m2[dst])));
}

__global__ void agg2_kernel(const int* __restrict__ ei, const float* __restrict__ als,
                            const float* __restrict__ ald, const unsigned* __restrict__ m2,
                            const float* __restrict__ den2, const float* __restrict__ h2,
                            float* __restrict__ agg2, int E, int Etot) {
  long long t = (long long)blockIdx.x * blockDim.x + threadIdx.x;
  if (t >= (long long)Etot * NCLS) return;
  int e   = (int)(t >> 4);
  int col = (int)(t & 15);
  int src = (e < E) ? ei[e]     : (e - E);
  int dst = (e < E) ? ei[E + e] : (e - E);
  float alpha = __expf(leaky(als[src] + ald[dst]) - dec_f32(m2[dst])) / den2[dst];
  atomicAdd(&agg2[(size_t)dst * NCLS + col], h2[(size_t)src * NCLS + col] * alpha);
}

// ---------------- bias + log_softmax ----------------
__global__ void final_kernel(const float* __restrict__ agg2, const float* __restrict__ b2,
                             float* __restrict__ out, int N) {
  int n = blockIdx.x * blockDim.x + threadIdx.x;
  if (n >= N) return;
  float v[NCLS];
  float mx = -3.4e38f;
  #pragma unroll
  for (int c = 0; c < NCLS; ++c) {
    v[c] = agg2[(size_t)n * NCLS + c] + b2[c];
    mx = fmaxf(mx, v[c]);
  }
  float s = 0.f;
  #pragma unroll
  for (int c = 0; c < NCLS; ++c) s += __expf(v[c] - mx);
  float lse = mx + __logf(s);
  #pragma unroll
  for (int c = 0; c < NCLS; ++c) out[(size_t)n * NCLS + c] = v[c] - lse;
}

extern "C" void kernel_launch(void* const* d_in, const int* in_sizes, int n_in,
                              void* d_out, int out_size, void* d_ws, size_t ws_size,
                              hipStream_t stream) {
  const float* x   = (const float*)d_in[0];
  const int*   ei  = (const int*)  d_in[1];
  const float* W1  = (const float*)d_in[2];
  const float* a1s = (const float*)d_in[3];
  const float* a1d = (const float*)d_in[4];
  const float* b1  = (const float*)d_in[5];
  const float* W2  = (const float*)d_in[6];
  const float* a2s = (const float*)d_in[7];
  const float* a2d = (const float*)d_in[8];
  const float* b2  = (const float*)d_in[9];

  const int N    = in_sizes[0] / F_IN;
  const int E    = in_sizes[1] / 2;
  const int Etot = E + N;

  float* ws = (float*)d_ws;
  size_t off = 0;
  __bf16* W1sw = (__bf16*)(ws + off); off += (8 * 4 * 32 * 16) / 2;  // 16384 bf16
  __bf16* W2sw = (__bf16*)(ws + off); off += (2 * 32 * 16) / 2;      // 1024 bf16
  float*    h1   = ws + off; off += (size_t)N * HC;
  float*    al1s = ws + off; off += (size_t)N * HEADS;
  float*    al1d = ws + off; off += (size_t)N * HEADS;
  float*    h2   = ws + off; off += (size_t)N * NCLS;
  float*    al2s = ws + off; off += (size_t)N;
  float*    al2d = ws + off; off += (size_t)N;
  float*    zb   = ws + off;                       // zero-init region start
  unsigned* m1   = (unsigned*)(ws + off); off += (size_t)N * HEADS;
  float*    den1 = ws + off; off += (size_t)N * HEADS;
  float*    agg1 = ws + off; off += (size_t)N * HC;
  unsigned* m2   = (unsigned*)(ws + off); off += (size_t)N;
  float*    den2 = ws + off; off += (size_t)N;
  float*    agg2 = ws + off; off += (size_t)N * NCLS;
  const size_t zcount = (size_t)N * (HEADS + HEADS + HC + 1 + 1 + NCLS);

  zero_kernel<<<(unsigned)((zcount + 255) / 256), 256, 0, stream>>>(zb, zcount);
  prep_w_kernel<<<64, 256, 0, stream>>>(W1, W2, W1sw, W2sw);

  gemm1_kernel<<<(N + 63) / 64, 128, 0, stream>>>(x, W1sw, a1s, a1d, h1, al1s, al1d, N);

  const unsigned gE = (Etot + 255) / 256;
  attmax1_kernel<<<gE, 256, 0, stream>>>(ei, al1s, al1d, m1, E, Etot);
  denom1_kernel <<<gE, 256, 0, stream>>>(ei, al1s, al1d, m1, den1, E, Etot);
  const long long t1 = (long long)Etot * HC;
  agg1_kernel<<<(unsigned)((t1 + 255) / 256), 256, 0, stream>>>(
      ei, al1s, al1d, m1, den1, h1, agg1, E, Etot);

  gemm2_kernel<<<(N + 63) / 64, 128, 0, stream>>>(agg1, b1, W2sw, a2s, a2d, h2, al2s, al2d, N);

  attmax2_kernel<<<gE, 256, 0, stream>>>(ei, al2s, al2d, m2, E, Etot);
  denom2_kernel <<<gE, 256, 0, stream>>>(ei, al2s, al2d, m2, den2, E, Etot);
  const long long t2 = (long long)Etot * NCLS;
  agg2_kernel<<<(unsigned)((t2 + 255) / 256), 256, 0, stream>>>(
      ei, al2s, al2d, m2, den2, h2, agg2, E, Etot);

  final_kernel<<<(N + 255) / 256, 256, 0, stream>>>(agg2, b2, (float*)d_out, N);
}
